// ParallelRetention_34660386079007
// MI455X (gfx1250) — matrix operand order repaired and checked by reading, not compiled, not run on hardware
//
#include <hip/hip_runtime.h>

// ---------------------------------------------------------------------------
// ParallelRetention for MI455X (gfx1250, wave32, WMMA, async global->LDS).
// f32->bf16 convert | fused QKV GEMM (double-buffered async LDS staging,
// wmma bf16 f32-acc, A tile shared across Wq/Wk/Wv) | kv·D^T GEMM fused with
// ⊙Q + per-block stats | deterministic stats reduce | layernorm write.
// ---------------------------------------------------------------------------

typedef __bf16 bf16;
typedef __attribute__((ext_vector_type(4)))  __bf16 v4bf;
typedef __attribute__((ext_vector_type(8)))  __bf16 v8bf;
typedef __attribute__((ext_vector_type(16))) __bf16 v16bf;
typedef __attribute__((ext_vector_type(8)))  float  v8f;
typedef __attribute__((ext_vector_type(4)))  int    v4i;

#define TPB   256
#define BK    64
#define LDSTR 80   // LDS row stride (bf16) for 64-wide K tiles: 160B, keeps 32B frag align

#if defined(__has_builtin)
#if __has_builtin(__builtin_amdgcn_global_load_async_to_lds_b128) && \
    __has_builtin(__builtin_amdgcn_s_wait_asynccnt)
#define USE_ASYNC 1
#endif
#endif
#ifndef USE_ASYNC
#define USE_ASYNC 0
#endif

#define AS1 __attribute__((address_space(1)))
#define AS3 __attribute__((address_space(3)))

union FragU { v16bf v; v8bf h[2]; };

static __device__ __forceinline__ v16bf make_frag(v8bf lo, v8bf hi) {
  FragU u; u.h[0] = lo; u.h[1] = hi; return u.v;
}

// Copy one 16B chunk global->LDS (async on CDNA5, sync fallback otherwise).
static __device__ __forceinline__ void cp16(bf16* l, const bf16* g) {
#if USE_ASYNC
  __builtin_amdgcn_global_load_async_to_lds_b128((AS1 v4i*)g, (AS3 v4i*)l, 0, 0);
#else
  *(v8bf*)l = *(const v8bf*)g;
#endif
}

static __device__ __forceinline__ void wait_async_le(int) {}
#if USE_ASYNC
#define WAIT_ASYNC(n) __builtin_amdgcn_s_wait_asynccnt(n)
#else
#define WAIT_ASYNC(n) wait_async_le(n)
#endif

// --------------------------- f32 -> bf16 convert ---------------------------
__global__ __launch_bounds__(TPB)
void cvt_f32_bf16(const float* __restrict__ src, bf16* __restrict__ dst) {
  const int i = blockIdx.x * TPB + threadIdx.x;      // one float4 per thread
  float4 v = ((const float4*)src)[i];
  v4bf o;
  o[0] = (bf16)v.x; o[1] = (bf16)v.y; o[2] = (bf16)v.z; o[3] = (bf16)v.w;
  ((v4bf*)dst)[i] = o;
}

// ------------------- fused Q/K/V projection (X · W^T + b) ------------------
// 64x64 C tile/block, 8 waves (2x4), each wave: two 16x16 tiles per matrix.
// Double-buffered LDS; async copies for stage k+1 overlap WMMA on stage k.
__global__ __launch_bounds__(TPB)
void qkv_gemm(const bf16* __restrict__ xb,
              const bf16* __restrict__ wqb, const bf16* __restrict__ wkb,
              const bf16* __restrict__ wvb,
              const float* __restrict__ bq, const float* __restrict__ bk,
              const float* __restrict__ bv,
              bf16* __restrict__ Qb, bf16* __restrict__ KVb) {
  constexpr int K = 2048;
  constexpr int N = 1024;
  __shared__ __align__(32) bf16 As [2][64 * LDSTR];
  __shared__ __align__(32) bf16 Bsq[2][64 * LDSTR];
  __shared__ __align__(32) bf16 Bsk[2][64 * LDSTR];
  __shared__ __align__(32) bf16 Bsv[2][64 * LDSTR];

  const int t    = threadIdx.x;
  const int m0   = blockIdx.y * 64;
  const int n0   = blockIdx.x * 64;
  const int lane = t & 31;
  const int wave = t >> 5;
  const int wm   = (wave & 1) * 32;
  const int wn   = (wave >> 1) * 16;
  const int fr   = lane & 15;
  const int kh   = lane >> 4;

  // Tile copy plan: 64 rows x 64 cols bf16 = 512 x 16B chunks; 2 per thread.
  int grow[2], gcol[2], loff[2];
#pragma unroll
  for (int j = 0; j < 2; ++j) {
    const int c = t + TPB * j;
    grow[j] = c >> 3;
    gcol[j] = (c & 7) << 3;
    loff[j] = grow[j] * LDSTR + gcol[j];
  }
  const bf16* ag = xb  + (size_t)m0 * K;
  const bf16* qg = wqb + (size_t)n0 * K;
  const bf16* kg = wkb + (size_t)n0 * K;
  const bf16* vg = wvb + (size_t)n0 * K;

  auto issue_stage = [&](int buf, int k0) {
#pragma unroll
    for (int j = 0; j < 2; ++j) {
      const size_t go = (size_t)grow[j] * K + k0 + gcol[j];
      cp16(&As [buf][loff[j]], ag + go);
      cp16(&Bsq[buf][loff[j]], qg + go);
      cp16(&Bsk[buf][loff[j]], kg + go);
      cp16(&Bsv[buf][loff[j]], vg + go);
    }
  };

  const v8f vzero = {0.f, 0.f, 0.f, 0.f, 0.f, 0.f, 0.f, 0.f};
  v8f accQ[2], accK[2], accV[2];
  accQ[0] = accQ[1] = accK[0] = accK[1] = accV[0] = accV[1] = vzero;

  issue_stage(0, 0);
  int cur = 0;
  for (int k0 = 0; k0 < K; k0 += BK) {
    __syncthreads();                       // prior stage's LDS reads retired
    const int kn = (k0 + BK < K) ? k0 + BK : k0;   // tail: benign re-issue
    issue_stage(cur ^ 1, kn);
    WAIT_ASYNC(8);                         // current buffer's 8 copies done
    __syncthreads();                       // visible to all waves

    const bf16* as = As [cur];
    const bf16* qs = Bsq[cur];
    const bf16* ks = Bsk[cur];
    const bf16* vs = Bsv[cur];
#pragma unroll
    for (int kk = 0; kk < BK; kk += 32) {
      const int brow = (wn + fr) * LDSTR + kk + kh * 16;
      const v16bf fq = *(const v16bf*)&qs[brow];
      const v16bf fk = *(const v16bf*)&ks[brow];
      const v16bf fv = *(const v16bf*)&vs[brow];
      v16bf fa[2];
#pragma unroll
      for (int s = 0; s < 2; ++s) {
        const bf16* ab = &as[(wm + s * 16 + fr) * LDSTR + kk + kh * 16];
        fa[s] = make_frag(*(const v8bf*)ab, *(const v8bf*)(ab + 8));
      }
#pragma unroll
      for (int s = 0; s < 2; ++s) {
        accQ[s] = __builtin_amdgcn_wmma_f32_16x16x32_bf16(false, fa[s], false, fq,
                                                          (short)0, accQ[s], false, false);
        accK[s] = __builtin_amdgcn_wmma_f32_16x16x32_bf16(false, fa[s], false, fk,
                                                          (short)0, accK[s], false, false);
        accV[s] = __builtin_amdgcn_wmma_f32_16x16x32_bf16(false, fa[s], false, fv,
                                                          (short)0, accV[s], false, false);
      }
    }
    cur ^= 1;
  }

  // C/D layout: column = fr, row = kh*8 + r within each 16x16 tile.
  const int ncol = n0 + wn + fr;
  const float bqv = bq[ncol], bkv = bk[ncol], bvv = bv[ncol];
#pragma unroll
  for (int s = 0; s < 2; ++s) {
    const int mbase = m0 + wm + s * 16 + kh * 8;
#pragma unroll
    for (int r = 0; r < 8; ++r) {
      const size_t o = (size_t)(mbase + r) * N + ncol;
      Qb[o]  = (bf16)(accQ[s][r] + bqv);
      KVb[o] = (bf16)((accK[s][r] + bkv) * (accV[s][r] + bvv));
    }
  }
}

// ---------------- out2 = kv · D^T, fused ⊙Q and block statistics -----------
__global__ __launch_bounds__(TPB)
void retention_gemm(const bf16* __restrict__ kvb, const bf16* __restrict__ db,
                    const bf16* __restrict__ Qb,
                    float* __restrict__ raw, float* __restrict__ part) {
  constexpr int K = 1024;
  constexpr int N = 1024;
  __shared__ __align__(32) bf16 As[2][64 * LDSTR];
  __shared__ __align__(32) bf16 Bs[2][64 * LDSTR];
  __shared__ float red[8][2];

  const int t    = threadIdx.x;
  const int m0   = blockIdx.y * 64;
  const int n0   = blockIdx.x * 64;
  const int lane = t & 31;
  const int wave = t >> 5;
  const int wm   = (wave & 1) * 32;
  const int wn   = (wave >> 1) * 16;
  const int fr   = lane & 15;
  const int kh   = lane >> 4;

  int grow[2], gcol[2], loff[2];
#pragma unroll
  for (int j = 0; j < 2; ++j) {
    const int c = t + TPB * j;
    grow[j] = c >> 3;
    gcol[j] = (c & 7) << 3;
    loff[j] = grow[j] * LDSTR + gcol[j];
  }
  const bf16* ag = kvb + (size_t)m0 * K;
  const bf16* bg = db  + (size_t)n0 * K;   // B[k][q] = D[q][k]

  auto issue_stage = [&](int buf, int k0) {
#pragma unroll
    for (int j = 0; j < 2; ++j) {
      const size_t go = (size_t)grow[j] * K + k0 + gcol[j];
      cp16(&As[buf][loff[j]], ag + go);
      cp16(&Bs[buf][loff[j]], bg + go);
    }
  };

  const v8f vzero = {0.f, 0.f, 0.f, 0.f, 0.f, 0.f, 0.f, 0.f};
  v8f acc[2];
  acc[0] = acc[1] = vzero;

  issue_stage(0, 0);
  int cur = 0;
  for (int k0 = 0; k0 < K; k0 += BK) {
    __syncthreads();
    const int kn = (k0 + BK < K) ? k0 + BK : k0;
    issue_stage(cur ^ 1, kn);
    WAIT_ASYNC(4);
    __syncthreads();

    const bf16* as = As[cur];
    const bf16* bs = Bs[cur];
#pragma unroll
    for (int kk = 0; kk < BK; kk += 32) {
      const v16bf fb = *(const v16bf*)&bs[(wn + fr) * LDSTR + kk + kh * 16];
      v16bf fa[2];
#pragma unroll
      for (int s = 0; s < 2; ++s) {
        const bf16* ab = &as[(wm + s * 16 + fr) * LDSTR + kk + kh * 16];
        fa[s] = make_frag(*(const v8bf*)ab, *(const v8bf*)(ab + 8));
      }
#pragma unroll
      for (int s = 0; s < 2; ++s)
        acc[s] = __builtin_amdgcn_wmma_f32_16x16x32_bf16(false, fa[s], false, fb,
                                                         (short)0, acc[s], false, false);
    }
    cur ^= 1;
  }

  // val = Q ⊙ (kv·D^T); write raw, accumulate sum / sumsq.
  float lsum = 0.f, lsq = 0.f;
  const int ncol = n0 + wn + fr;
#pragma unroll
  for (int s = 0; s < 2; ++s) {
    const int mbase = m0 + wm + s * 16 + kh * 8;
#pragma unroll
    for (int r = 0; r < 8; ++r) {
      const size_t o = (size_t)(mbase + r) * N + ncol;
      const float val = (float)Qb[o] * acc[s][r];
      raw[o] = val;
      lsum += val;
      lsq  += val * val;
    }
  }
#pragma unroll
  for (int off = 16; off > 0; off >>= 1) {
    lsum += __shfl_xor(lsum, off, 32);
    lsq  += __shfl_xor(lsq,  off, 32);
  }
  if (lane == 0) { red[wave][0] = lsum; red[wave][1] = lsq; }
  __syncthreads();
  if (t == 0) {
    float s = 0.f, q = 0.f;
#pragma unroll
    for (int w = 0; w < 8; ++w) { s += red[w][0]; q += red[w][1]; }
    const int p = blockIdx.y * gridDim.x + blockIdx.x;  // 256 partials per batch
    part[2 * p]     = s;
    part[2 * p + 1] = q;
  }
}

// ------------------- deterministic per-batch stats reduce ------------------
__global__ __launch_bounds__(TPB)
void reduce_stats(const float* __restrict__ part, float* __restrict__ stats) {
  __shared__ float ss[TPB], sq[TPB];
  const int b = blockIdx.x, t = threadIdx.x;
  ss[t] = part[2 * (b * 256 + t)];
  sq[t] = part[2 * (b * 256 + t) + 1];
  __syncthreads();
  for (int off = 128; off > 0; off >>= 1) {
    if (t < off) { ss[t] += ss[t + off]; sq[t] += sq[t + off]; }
    __syncthreads();
  }
  if (t == 0) { stats[2 * b] = ss[0]; stats[2 * b + 1] = sq[0]; }
}

// ------------------------------- layernorm ---------------------------------
__global__ __launch_bounds__(TPB)
void norm_write(const float* __restrict__ raw, const float* __restrict__ stats,
                float* __restrict__ out) {
  const int i = blockIdx.x * TPB + threadIdx.x;   // one float4 per thread
  const int b = i >> 18;                          // (i*4) >> 20
  const float inv  = 1.0f / (1024.0f * 1024.0f);
  const float mean = stats[2 * b] * inv;
  float var = fmaf(-mean, mean, stats[2 * b + 1] * inv);
  var = var < 0.f ? 0.f : var;
  const float rs = rsqrtf(var + 1e-5f);
  float4 v = ((const float4*)raw)[i];
  v.x = (v.x - mean) * rs;
  v.y = (v.y - mean) * rs;
  v.z = (v.z - mean) * rs;
  v.w = (v.w - mean) * rs;
  ((float4*)out)[i] = v;
}

// ------------------------------- launcher ----------------------------------
extern "C" void kernel_launch(void* const* d_in, const int* in_sizes, int n_in,
                              void* d_out, int out_size, void* d_ws, size_t ws_size,
                              hipStream_t stream) {
  (void)in_sizes; (void)n_in; (void)out_size; (void)ws_size;
  const float* x  = (const float*)d_in[0];
  const float* Wq = (const float*)d_in[1];
  const float* bq = (const float*)d_in[2];
  const float* Wk = (const float*)d_in[3];
  const float* bk = (const float*)d_in[4];
  const float* Wv = (const float*)d_in[5];
  const float* bv = (const float*)d_in[6];
  const float* dm = (const float*)d_in[7];   // (1,1024,1024) decay mask
  float* out = (float*)d_out;

  constexpr size_t M = 8192, N = 1024, K = 2048;
  char* ws = (char*)d_ws;
  size_t off = 0;
  auto alloc = [&](size_t bytes) -> char* {
    char* p = ws + off;
    off += (bytes + 255) & ~(size_t)255;
    return p;
  };
  bf16*  xb    = (bf16*)alloc(M * K * 2);       // 32 MB
  bf16*  wqb   = (bf16*)alloc(N * K * 2);       // 4 MB
  bf16*  wkb   = (bf16*)alloc(N * K * 2);       // 4 MB
  bf16*  wvb   = (bf16*)alloc(N * K * 2);       // 4 MB
  bf16*  db    = (bf16*)alloc(N * N * 2);       // 2 MB
  bf16*  Qb    = (bf16*)alloc(M * N * 2);       // 16 MB
  bf16*  KVb   = (bf16*)alloc(M * N * 2);       // 16 MB
  float* raw   = (float*)alloc(M * N * 4);      // 32 MB
  float* part  = (float*)alloc(2048 * 2 * 4);
  float* stats = (float*)alloc(16 * 4);

  cvt_f32_bf16<<<dim3((unsigned)(M * K / 1024)), TPB, 0, stream>>>(x,  xb);
  cvt_f32_bf16<<<dim3((unsigned)(N * K / 1024)), TPB, 0, stream>>>(Wq, wqb);
  cvt_f32_bf16<<<dim3((unsigned)(N * K / 1024)), TPB, 0, stream>>>(Wk, wkb);
  cvt_f32_bf16<<<dim3((unsigned)(N * K / 1024)), TPB, 0, stream>>>(Wv, wvb);
  cvt_f32_bf16<<<dim3((unsigned)(N * N / 1024)), TPB, 0, stream>>>(dm, db);

  dim3 gtile((unsigned)(N / 64), (unsigned)(M / 64));   // 16 x 128 blocks
  qkv_gemm<<<gtile, TPB, 0, stream>>>(xb, wqb, wkb, wvb, bq, bk, bv, Qb, KVb);
  retention_gemm<<<gtile, TPB, 0, stream>>>(KVb, db, Qb, raw, part);
  reduce_stats<<<dim3(8), TPB, 0, stream>>>(part, stats);
  norm_write<<<dim3((unsigned)(M * N / 4 / TPB)), TPB, 0, stream>>>(raw, stats, out);
}
